// GPT2Attention_32547262169547
// MI455X (gfx1250) — compile-verified, matrix operand order checked by reading, not run on hardware
//
#include <hip/hip_runtime.h>

// ---------------------------------------------------------------------------
// GPT-2 attention block for gfx1250 (MI455X), bf16 WMMA pipeline with
// async global->LDS double buffering.  B=4, S=2048, D=1024, H=16, hd=64.
// ---------------------------------------------------------------------------

typedef __attribute__((ext_vector_type(16))) __bf16 v16bf;
typedef __attribute__((ext_vector_type(8)))  float  v8f;

union FragU { unsigned int u[8]; v16bf bf; };
union FragF { float f[8]; v8f v; };

__device__ __forceinline__ unsigned short f2bf(float x) {
    unsigned int u = __float_as_uint(x);
    u += 0x7FFFu + ((u >> 16) & 1u);          // round-to-nearest-even
    return (unsigned short)(u >> 16);
}

// generic (shared) pointer -> 32-bit LDS byte offset (low 32 bits of the address)
__device__ __forceinline__ unsigned lds_off(const void* p) {
    return (unsigned)(unsigned long long)p;
}

// one 16-byte async global->LDS copy per lane (ASYNCcnt-tracked)
__device__ __forceinline__ void async_b128(unsigned lds, unsigned long long base, unsigned goff) {
    asm volatile("global_load_async_to_lds_b128 %0, %1, %2"
                 :: "v"(lds), "v"(goff), "s"(base) : "memory");
}
// 32-byte copy (two b128)
__device__ __forceinline__ void async_b256(unsigned lds, unsigned long long base, unsigned goff) {
    async_b128(lds, base, goff);
    async_b128(lds + 16u, base, goff + 16u);
}

// Load a 16x32 bf16 fragment (A-layout == B-column-major layout) from LDS.
// Lane L: row = L%16; v0..v3: K = 8*(L/16)+{0..7}; v4..v7: K = 16+8*(L/16)+{0..7}
__device__ __forceinline__ v16bf ld_frag_b16(const unsigned short* base, int stride, int coloff) {
    const int lane = threadIdx.x & 31;
    const int row  = lane & 15;
    const int kb   = (lane >> 4) << 3;
    const unsigned short* p = base + row * stride + coloff + kb;
    FragU f;
#pragma unroll
    for (int v = 0; v < 4; ++v) {
        f.u[v]     = *(const unsigned int*)(p + 2 * v);
        f.u[v + 4] = *(const unsigned int*)(p + 16 + 2 * v);
    }
    return f.bf;
}

__device__ __forceinline__ v8f wmma_bf16(v16bf a, v16bf b, v8f c) {
    return __builtin_amdgcn_wmma_f32_16x16x32_bf16(false, a, false, b, (short)0, c, false, false);
}

// ---------------------------------------------------------------------------
// fp32 -> bf16 conversion (row-major copy)
// ---------------------------------------------------------------------------
__global__ void cvt_f32_bf16(const float* __restrict__ in, unsigned short* __restrict__ out, int n) {
    int i = blockIdx.x * blockDim.x + threadIdx.x;
    if (i < n) out[i] = f2bf(in[i]);
}

// ---------------------------------------------------------------------------
// fp32 [R][C] -> bf16 [C][R] LDS-tiled transpose-convert (weights, one-time)
// block (32,8), grid (C/32, R/32)
// ---------------------------------------------------------------------------
__global__ __launch_bounds__(256) void cvt_transpose_bf16(
    const float* __restrict__ in, unsigned short* __restrict__ out, int R, int C)
{
    __shared__ float tile[32][33];
    const int c0 = blockIdx.x << 5, r0 = blockIdx.y << 5;
    const int tx = threadIdx.x, ty = threadIdx.y;
#pragma unroll
    for (int j = 0; j < 4; ++j)
        tile[ty + j * 8][tx] = in[(size_t)(r0 + ty + j * 8) * C + c0 + tx];
    __syncthreads();
#pragma unroll
    for (int j = 0; j < 4; ++j)
        out[(size_t)(c0 + ty + j * 8) * R + r0 + tx] = f2bf(tile[tx][ty + j * 8]);
}

// ---------------------------------------------------------------------------
// Tiled bf16 GEMM: C[M,N] = A[M,K] @ Bt[N,K]^T + bias.
// Block tile 128x128, 8 waves of 64x32, K-step 32, async double-buffered LDS.
// MODE 0: scatter +bias into Q/K [B,H,S,64] and V transposed [B,H,64,S] (bf16)
// MODE 1: store +bias as f32 row-major (output projection)
// ---------------------------------------------------------------------------
template <int MODE>
__global__ __launch_bounds__(256) void gemm_bf16(
    const unsigned short* __restrict__ A, const unsigned short* __restrict__ Bt,
    const float* __restrict__ bias,
    unsigned short* outQ, unsigned short* outK, unsigned short* outV,
    float* outF, int M, int N, int K)
{
    __shared__ unsigned short As[2][128 * 40];   // 128 m x 32 k, stride 40
    __shared__ unsigned short Bs[2][128 * 40];   // 128 n x 32 k, stride 40

    const int tid  = threadIdx.x;
    const int lane = tid & 31;
    const int wid  = tid >> 5;
    const int m0   = blockIdx.y << 7;
    const int n0   = blockIdx.x << 7;
    const int wm   = wid & 1;
    const int wn   = wid >> 1;

    const int row  = tid >> 1, hseg = tid & 1;   // 16 elems (32B) per thread per tile
    const unsigned long long baseA = (unsigned long long)A;
    const unsigned long long baseB = (unsigned long long)Bt;

    const v8f vzero = {0.f, 0.f, 0.f, 0.f, 0.f, 0.f, 0.f, 0.f};
    v8f acc[4][2];
#pragma unroll
    for (int mt = 0; mt < 4; ++mt) { acc[mt][0] = vzero; acc[mt][1] = vzero; }

    // per tile: 2 asyncs for A + 2 for B = 4 per thread
    const unsigned ldsA0 = lds_off(&As[0][row * 40 + hseg * 16]);
    const unsigned ldsB0 = lds_off(&Bs[0][row * 40 + hseg * 16]);
    const unsigned bufstep = (unsigned)(128 * 40 * sizeof(unsigned short));
    const unsigned goffA0 = (unsigned)(((size_t)(m0 + row) * K + hseg * 16) * 2);
    const unsigned goffB0 = (unsigned)(((size_t)(n0 + row) * K + hseg * 16) * 2);

    // preload tile 0
    async_b256(ldsA0, baseA, goffA0);
    async_b256(ldsB0, baseB, goffB0);

    const int nk = K >> 5;
    for (int t = 0; t < nk; ++t) {
        const int cur = t & 1;
        if (t + 1 < nk) {
            const unsigned koff = (unsigned)((t + 1) << 6);   // 32 elems * 2B
            async_b256(ldsA0 + (1 - cur) * bufstep, baseA, goffA0 + koff);
            async_b256(ldsB0 + (1 - cur) * bufstep, baseB, goffB0 + koff);
            asm volatile("s_wait_asynccnt 4" ::: "memory");   // tile t landed
        } else {
            asm volatile("s_wait_asynccnt 0" ::: "memory");
        }
        __syncthreads();

        v16bf af[4];
#pragma unroll
        for (int mt = 0; mt < 4; ++mt)
            af[mt] = ld_frag_b16(&As[cur][(wm * 64 + mt * 16) * 40], 40, 0);
        v16bf bf0 = ld_frag_b16(&Bs[cur][(wn * 32) * 40], 40, 0);
        v16bf bf1 = ld_frag_b16(&Bs[cur][(wn * 32 + 16) * 40], 40, 0);
#pragma unroll
        for (int mt = 0; mt < 4; ++mt) {
            acc[mt][0] = wmma_bf16(af[mt], bf0, acc[mt][0]);
            acc[mt][1] = wmma_bf16(af[mt], bf1, acc[mt][1]);
        }
        __syncthreads();   // buffer (cur) reusable at t+2
    }

    // ---- epilogue ----
    const int half = lane >> 4;
    const int nc   = lane & 15;
#pragma unroll
    for (int mt = 0; mt < 4; ++mt) {
#pragma unroll
        for (int nt = 0; nt < 2; ++nt) {
            const int col = n0 + wn * 32 + nt * 16 + nc;
            const float bv = bias[col];
            FragF f; f.v = acc[mt][nt];
            const int mbase = m0 + wm * 64 + mt * 16 + 8 * half;
            if (MODE == 0) {
                const int which = col >> 10;       // 0=q 1=k 2=v
                const int d = col & 1023;
                const int h = d >> 6, c = d & 63;
                unsigned short* dst = (which == 0) ? outQ : ((which == 1) ? outK : outV);
#pragma unroll
                for (int v = 0; v < 8; ++v) {
                    const int m = mbase + v;
                    const int b = m >> 11, s = m & 2047;
                    const size_t idx = (which == 2)
                        ? ((size_t)(b * 16 + h) * 64 + c) * 2048 + s      // V transposed [B,H,64,S]
                        : ((size_t)(b * 16 + h) * 2048 + s) * 64 + c;    // Q/K [B,H,S,64]
                    dst[idx] = f2bf(f.f[v] + bv);
                }
            } else {
#pragma unroll
                for (int v = 0; v < 8; ++v) {
                    const int m = mbase + v;
                    outF[(size_t)m * N + col] = f.f[v] + bv;
                }
            }
        }
    }
}

// ---------------------------------------------------------------------------
// Causal flash attention per (b,h, 128-query tile). 8 waves x 16 query rows.
// K/Vt tiles (32 keys) async double-buffered through LDS; softmax in fp32.
// ---------------------------------------------------------------------------
__global__ __launch_bounds__(256) void attn_kernel(
    const unsigned short* __restrict__ qb, const unsigned short* __restrict__ kb,
    const unsigned short* __restrict__ vtb, unsigned short* __restrict__ ob)
{
    __shared__ unsigned short Qs[128 * 72];        // 128 q-rows x 64 ch, stride 72
    __shared__ unsigned short Ks[2][32 * 72];      // 32 keys x 64 ch
    __shared__ unsigned short Vts[2][64 * 40];     // 64 ch x 32 keys (already transposed)
    __shared__ unsigned short Ps[8 * 16 * 40];     // per-wave P staging

    const int bh  = blockIdx.x >> 4;               // b*16+h
    const int q0  = (blockIdx.x & 15) << 7;
    const int tid = threadIdx.x;
    const int lane = tid & 31, wid = tid >> 5;
    const int half = lane >> 4, nc = lane & 15;
    const unsigned long long baseQ = (unsigned long long)(qb  + (size_t)bh * 2048 * 64);
    const unsigned long long baseK = (unsigned long long)(kb  + (size_t)bh * 2048 * 64);
    const unsigned long long baseV = (unsigned long long)(vtb + (size_t)bh * 64 * 2048);

    {   // stage Q tile: 4 async b128 per thread
        const int qrow = tid >> 1, qseg = tid & 1;
        const unsigned lq = lds_off(&Qs[qrow * 72 + qseg * 32]);
        const unsigned gq = (unsigned)(((size_t)(q0 + qrow) * 64 + qseg * 32) * 2);
#pragma unroll
        for (int j = 0; j < 4; ++j) async_b128(lq + j * 16u, baseQ, gq + j * 16u);
    }

    // K tile: thread -> (key row, 8-ch seg); Vt tile: thread -> (channel, 8-key seg)
    const int krow = tid >> 3, kseg = tid & 7;
    const int vch  = tid >> 2, vseg = tid & 3;
    const unsigned ldsK0 = lds_off(&Ks[0][krow * 72 + kseg * 8]);
    const unsigned ldsV0 = lds_off(&Vts[0][vch * 40 + vseg * 8]);
    const unsigned kbufstep = (unsigned)(32 * 72 * sizeof(unsigned short));
    const unsigned vbufstep = (unsigned)(64 * 40 * sizeof(unsigned short));
    const unsigned goffK0 = (unsigned)(((size_t)krow * 64 + kseg * 8) * 2);
    const unsigned goffV0 = (unsigned)(((size_t)vch * 2048 + vseg * 8) * 2);

    // preload K/V tile 0 (2 asyncs per thread per tile)
    async_b128(ldsK0, baseK, goffK0);
    async_b128(ldsV0, baseV, goffV0);

    asm volatile("s_wait_asynccnt 2" ::: "memory");   // Q landed (K/V0 may be in flight)
    __syncthreads();

    const v16bf qf0 = ld_frag_b16(&Qs[(wid * 16) * 72], 72, 0);
    const v16bf qf1 = ld_frag_b16(&Qs[(wid * 16) * 72], 72, 32);

    const v8f vzero = {0.f, 0.f, 0.f, 0.f, 0.f, 0.f, 0.f, 0.f};
    v8f o[4];
#pragma unroll
    for (int ct = 0; ct < 4; ++ct) o[ct] = vzero;
    float mrow[8], lrow[8];
#pragma unroll
    for (int v = 0; v < 8; ++v) { mrow[v] = -1e30f; lrow[v] = 0.f; }

    const int qrow_base = q0 + wid * 16 + 8 * half;
    const int nk = (q0 >> 5) + 4;                     // key tiles through diagonal block
    unsigned short* Pw = &Ps[wid * 16 * 40];

    for (int t = 0; t < nk; ++t) {
        const int k0 = t << 5;
        const int cur = t & 1;
        if (t + 1 < nk) {
            const unsigned row64 = (unsigned)((t + 1) << 12);   // 32 keys * 64ch * 2B
            const unsigned col32 = (unsigned)((t + 1) << 6);    // 32 keys * 2B
            async_b128(ldsK0 + (1 - cur) * kbufstep, baseK, goffK0 + row64);
            async_b128(ldsV0 + (1 - cur) * vbufstep, baseV, goffV0 + col32);
            asm volatile("s_wait_asynccnt 2" ::: "memory");     // tile t landed
        } else {
            asm volatile("s_wait_asynccnt 0" ::: "memory");
        }
        __syncthreads();

        // scores: two 16x16 key tiles, head-dim 64 = 2 chained WMMAs each
        v8f sc0 = vzero, sc1 = vzero;
        sc0 = wmma_bf16(qf0, ld_frag_b16(&Ks[cur][0], 72, 0),  sc0);
        sc0 = wmma_bf16(qf1, ld_frag_b16(&Ks[cur][0], 72, 32), sc0);
        sc1 = wmma_bf16(qf0, ld_frag_b16(&Ks[cur][16 * 72], 72, 0),  sc1);
        sc1 = wmma_bf16(qf1, ld_frag_b16(&Ks[cur][16 * 72], 72, 32), sc1);

        FragF a0; a0.v = sc0;
        FragF a1; a1.v = sc1;
        float pm[8];
#pragma unroll
        for (int v = 0; v < 8; ++v) {
            const int q = qrow_base + v;
            float s0 = a0.f[v] * 0.125f;               // 1/sqrt(64)
            float s1 = a1.f[v] * 0.125f;
            if (k0 + nc      > q) s0 = -10000.f;       // causal mask bias
            if (k0 + 16 + nc > q) s1 = -10000.f;
            float mx = fmaxf(s0, s1);
#pragma unroll
            for (int off = 1; off < 16; off <<= 1) mx = fmaxf(mx, __shfl_xor(mx, off, 32));
            const float mn  = fmaxf(mrow[v], mx);
            const float scl = __expf(mrow[v] - mn);
            const float p0  = __expf(s0 - mn);
            const float p1  = __expf(s1 - mn);
            float sum = p0 + p1;
#pragma unroll
            for (int off = 1; off < 16; off <<= 1) sum += __shfl_xor(sum, off, 32);
            lrow[v] = lrow[v] * scl + sum;
            mrow[v] = mn;
            pm[v]   = scl;
            const int r = v + 8 * half;
            Pw[r * 40 + nc]      = f2bf(p0);
            Pw[r * 40 + 16 + nc] = f2bf(p1);
        }
#pragma unroll
        for (int ct = 0; ct < 4; ++ct) {               // rescale running output
            FragF f; f.v = o[ct];
#pragma unroll
            for (int v = 0; v < 8; ++v) f.f[v] *= pm[v];
            o[ct] = f.v;
        }
        asm volatile("s_wait_dscnt 0" ::: "memory");   // wave-local P visibility
        const v16bf pf = ld_frag_b16(Pw, 40, 0);
        o[0] = wmma_bf16(pf, ld_frag_b16(&Vts[cur][0],       40, 0), o[0]);
        o[1] = wmma_bf16(pf, ld_frag_b16(&Vts[cur][16 * 40], 40, 0), o[1]);
        o[2] = wmma_bf16(pf, ld_frag_b16(&Vts[cur][32 * 40], 40, 0), o[2]);
        o[3] = wmma_bf16(pf, ld_frag_b16(&Vts[cur][48 * 40], 40, 0), o[3]);
        __syncthreads();                               // buffer (cur) reusable at t+2
    }

    // finalize: divide by row sums, store merged-head bf16 [B*S, 1024]
    const int b = bh >> 4, h = bh & 15;
#pragma unroll
    for (int ct = 0; ct < 4; ++ct) {
        FragF f; f.v = o[ct];
#pragma unroll
        for (int v = 0; v < 8; ++v) {
            const int s = qrow_base + v;
            ob[((size_t)(b * 2048 + s)) * 1024 + h * 64 + ct * 16 + nc] =
                f2bf(f.f[v] / lrow[v]);
        }
    }
}

// ---------------------------------------------------------------------------
// Launcher
// ---------------------------------------------------------------------------
extern "C" void kernel_launch(void* const* d_in, const int* in_sizes, int n_in,
                              void* d_out, int out_size, void* d_ws, size_t ws_size,
                              hipStream_t stream) {
    (void)in_sizes; (void)n_in; (void)out_size; (void)ws_size;
    const float* hs     = (const float*)d_in[0];   // [4,2048,1024]
    const float* w_attn = (const float*)d_in[1];   // [1024,3072]
    const float* b_attn = (const float*)d_in[2];   // [3072]
    const float* w_proj = (const float*)d_in[3];   // [1024,1024]
    const float* b_proj = (const float*)d_in[4];   // [1024]
    float* out = (float*)d_out;                    // [4,2048,1024] fp32

    unsigned short* Xbf = (unsigned short*)d_ws;                 // 8192*1024
    unsigned short* WaT = Xbf + (size_t)8192 * 1024;             // [3072][1024] transposed
    unsigned short* WpT = WaT + (size_t)3072 * 1024;             // [1024][1024] transposed
    unsigned short* Qb  = WpT + (size_t)1024 * 1024;             // [64,2048,64]
    unsigned short* Kb  = Qb  + (size_t)64 * 2048 * 64;
    unsigned short* Vtb = Kb  + (size_t)64 * 2048 * 64;          // [64,64,2048] (B*H, hd, S)
    unsigned short* Ab  = Vtb + (size_t)64 * 2048 * 64;          // 8192*1024

    int n = 8192 * 1024;
    cvt_f32_bf16<<<(n + 255) / 256, 256, 0, stream>>>(hs, Xbf, n);

    dim3 tb(32, 8);
    cvt_transpose_bf16<<<dim3(3072 / 32, 1024 / 32), tb, 0, stream>>>(w_attn, WaT, 1024, 3072);
    cvt_transpose_bf16<<<dim3(1024 / 32, 1024 / 32), tb, 0, stream>>>(w_proj, WpT, 1024, 1024);

    dim3 gqkv(3072 / 128, 8192 / 128);
    gemm_bf16<0><<<gqkv, 256, 0, stream>>>(Xbf, WaT, b_attn, Qb, Kb, Vtb, nullptr,
                                           8192, 3072, 1024);

    attn_kernel<<<64 * 16, 256, 0, stream>>>(Qb, Kb, Vtb, Ab);

    dim3 gproj(1024 / 128, 8192 / 128);
    gemm_bf16<1><<<gproj, 256, 0, stream>>>(Ab, WpT, b_proj, nullptr, nullptr, nullptr, out,
                                            8192, 1024, 1024);
}
